// AlwaysOnMoeOnForward_94489280669
// MI455X (gfx1250) — matrix operand order repaired and checked by reading, not compiled, not run on hardware
//
#include <hip/hip_runtime.h>

// ---------------------------------------------------------------------------
// AlwaysOn MoE forward for MI455X (gfx1250, wave32, WMMA).
// bf16 WMMA GEMMs, f32 accumulation.
//  - A (activation) tiles:  GLOBAL_LOAD_ASYNC_TO_LDS_B128  (ASYNCcnt)
//  - B (weight) tiles:      TENSOR_LOAD_TO_LDS (TDM, D# descriptor, TENSORcnt)
//  - double-buffered LDS, ds_load_b128 fragments, contiguous k = 16*(lane/16) + s per lane
// ---------------------------------------------------------------------------

typedef __bf16 bf16_t;
typedef __attribute__((ext_vector_type(16))) __bf16 v16bf;
typedef __attribute__((ext_vector_type(8)))  float  v8f;
typedef __attribute__((ext_vector_type(4)))  unsigned int v4u;
typedef __attribute__((ext_vector_type(4)))  int v4i_t;
typedef __attribute__((ext_vector_type(8)))  int v8i_t;

union FragAB {           // 16 bf16 = 32 bytes = two 16B quads
    v16bf v;
    v4u   q[2];
};

#define T_TOK 4096
#define DIM   1024
#define DFFN  2048
#define NE    8

#if __has_builtin(__builtin_amdgcn_tensor_load_to_lds)
#define HAVE_TDM 1
#else
#define HAVE_TDM 0
#endif

// ---------------------------------------------------------------------------
// CDNA5 async copy global->LDS (ASYNCcnt) + waits
// ---------------------------------------------------------------------------
__device__ __forceinline__ void async_copy_b128(void* lds_dst, const void* gsrc)
{
#if __has_builtin(__builtin_amdgcn_global_load_async_to_lds_b128)
    __builtin_amdgcn_global_load_async_to_lds_b128(
        (__attribute__((address_space(1))) v4i_t*)gsrc,
        (__attribute__((address_space(3))) v4i_t*)lds_dst,
        0, 0);
#else
    const unsigned lo = (unsigned)(unsigned long long)lds_dst;
    asm volatile("global_load_async_to_lds_b128 %0, %1, off"
                 :: "v"(lo), "v"(gsrc)
                 : "memory");
#endif
}

template <int N>
__device__ __forceinline__ void wait_asynccnt()
{
#if __has_builtin(__builtin_amdgcn_s_wait_asynccnt)
    __builtin_amdgcn_s_wait_asynccnt(N);
#else
    asm volatile("s_wait_asynccnt %0" :: "i"(N) : "memory");
#endif
}

template <int N>
__device__ __forceinline__ void wait_tensorcnt()
{
#if __has_builtin(__builtin_amdgcn_s_wait_tensorcnt)
    __builtin_amdgcn_s_wait_tensorcnt(N);
#else
    asm volatile("s_wait_tensorcnt %0" :: "i"(N) : "memory");
#endif
}

__device__ __forceinline__ unsigned lds_off(const void* p)
{
    return (unsigned)(unsigned long long)p;   // wave-relative LDS byte address
}

#if HAVE_TDM
// 2D tile DMA: rows x cols bf16 elements, row stride `stride_elems`, packed
// contiguously into LDS at lds_addr. D# per ISA 8.3/8.4 (2D: groups 2/3 zero).
__device__ __forceinline__ void tdm_load_2d(unsigned lds_addr, const void* gsrc,
                                            int rows, int cols_elems,
                                            int stride_elems)
{
    const unsigned long long ga = (unsigned long long)gsrc;
    v4u g0;
    g0[0] = 1u;                                   // count=1, user mode
    g0[1] = lds_addr;                             // lds_addr[31:0]
    g0[2] = (unsigned)ga;                         // global_addr[31:0]
    g0[3] = (unsigned)((ga >> 32) & 0x01FFFFFFull) | 0x80000000u; // addr[56:32], type=2
    v8i_t g1;
    g1[0] = (int)(1u << 16);                      // wg_mask=0, data_size=1 (2B)
    g1[1] = 0;                                    // no atomic barrier; td0 lo16=0
    g1[2] = 0x4000;                               // tensor_dim0 = 1<<30 (no clip)
    g1[3] = 0x4000 | (cols_elems << 16);          // tensor_dim1 = 1<<30; tile_dim0
    g1[4] = rows;                                 // tile_dim1; tile_dim2=0 (2D)
    g1[5] = stride_elems;                         // tensor_dim0_stride[31:0]
    g1[6] = 0;
    g1[7] = 0;
    v4i_t gz  = {0, 0, 0, 0};
    v8i_t gz8 = {0, 0, 0, 0, 0, 0, 0, 0};
    // 6-arg toolchain form: (g0, g1, g2, g3, extra_v8i, cpol)
    __builtin_amdgcn_tensor_load_to_lds(g0, g1, gz, gz, gz8, 0);
}
#endif

// ---------------------------------------------------------------------------
// Router: one wave32 per token -> dense combine weights cw[T,7].
// ---------------------------------------------------------------------------
__global__ __launch_bounds__(256)
void router_kernel(const float* __restrict__ x,
                   const float* __restrict__ rw,
                   const float* __restrict__ rb,
                   float* __restrict__ cw)
{
    const int lane = threadIdx.x & 31;
    const int wid  = threadIdx.x >> 5;
    const int tok  = blockIdx.x * 8 + wid;
    const float* xr = x + (size_t)tok * DIM;

    float acc[7] = {0.f, 0.f, 0.f, 0.f, 0.f, 0.f, 0.f};
    for (int d = lane; d < DIM; d += 32) {
        const float xv = xr[d];
        const float* wr = rw + (size_t)d * 7;
#pragma unroll
        for (int e = 0; e < 7; ++e) acc[e] += xv * wr[e];
    }
#pragma unroll
    for (int e = 0; e < 7; ++e) {
        float v = acc[e];
#pragma unroll
        for (int off = 16; off > 0; off >>= 1) v += __shfl_xor(v, off, 32);
        acc[e] = v + rb[e];
    }
    if (lane == 0) {
        int i1 = 0; float l1 = acc[0];
#pragma unroll
        for (int e = 1; e < 7; ++e) if (acc[e] > l1) { l1 = acc[e]; i1 = e; }
        int i2 = -1; float l2 = -3.4e38f;
#pragma unroll
        for (int e = 0; e < 7; ++e) if (e != i1 && acc[e] > l2) { l2 = acc[e]; i2 = e; }
        const float p1 = 1.0f / (1.0f + __expf(l2 - l1));
        float* o = cw + (size_t)tok * 7;
#pragma unroll
        for (int e = 0; e < 7; ++e) o[e] = 0.0f;
        o[i1] = p1;
        o[i2] = 1.0f - p1;
    }
}

// ---------------------------------------------------------------------------
// f32 -> bf16 elementwise convert (activations)
// ---------------------------------------------------------------------------
__global__ __launch_bounds__(256)
void convert_x_kernel(const float* __restrict__ src, bf16_t* __restrict__ dst, int n)
{
    const int i = blockIdx.x * 256 + threadIdx.x;
    if (i < n) dst[i] = (bf16_t)src[i];
}

// ---------------------------------------------------------------------------
// Transpose + convert: src f32 [z][R][C] -> dst bf16 [z][C][R] (LDS 32x32 tile)
// ---------------------------------------------------------------------------
__global__ __launch_bounds__(256)
void transpose_convert_kernel(const float* __restrict__ src,
                              bf16_t* __restrict__ dst,
                              int R, int C)
{
    __shared__ float tile[32][33];
    const size_t plane = (size_t)R * C;
    const float* s = src + plane * blockIdx.z;
    bf16_t*      d = dst + plane * blockIdx.z;
    const int c0 = blockIdx.x * 32;
    const int r0 = blockIdx.y * 32;
#pragma unroll
    for (int j = 0; j < 32; j += 8)
        tile[threadIdx.y + j][threadIdx.x] =
            s[(size_t)(r0 + threadIdx.y + j) * C + (c0 + threadIdx.x)];
    __syncthreads();
#pragma unroll
    for (int j = 0; j < 32; j += 8)
        d[(size_t)(c0 + threadIdx.y + j) * R + (r0 + threadIdx.x)] =
            (bf16_t)tile[threadIdx.x][threadIdx.y + j];
}

// ---------------------------------------------------------------------------
// Fused gate+up GEMM + SwiGLU. A tiles via async-to-LDS, B tiles via TDM.
//   A  = xb  [T, D] bf16 ; Bg = wgt [DFF, D] ; Bu = wut [DFF, D]
//   H  = silu(A@Bg) * (A@Bu) -> bf16 [T, DFF]
// 256 thr = 8 waves (4M x 2N). Block tile 128x64, wave tile 32x32, k-step 32.
// ---------------------------------------------------------------------------
__global__ __launch_bounds__(256)
void gateup_kernel(const bf16_t* __restrict__ xb,
                   const bf16_t* __restrict__ wgt,
                   const bf16_t* __restrict__ wut,
                   bf16_t* __restrict__ hc)
{
    __shared__ bf16_t As[2][128][32];
    __shared__ bf16_t Gs[2][64][32];
    __shared__ bf16_t Us[2][64][32];

    const int lane  = threadIdx.x & 31;
    const int wave  = threadIdx.x >> 5;
    const int waveM = wave & 3;
    const int waveN = wave >> 2;
    const int llo   = lane & 15;
    const int lhi   = lane >> 4;

    const int rowBlk = blockIdx.y * 128;
    const int colBlk = blockIdx.x * 64;

    const int cr = lane >> 2;        // 0..7 row within 8-row slab
    const int cc = (lane & 3) * 8;   // bf16 element offset within row

    auto issueA = [&](int buf, int k) {
        const int ra0 = wave * 8 + cr;            // A rows 0..63
        async_copy_b128(&As[buf][ra0][cc],
                        xb + (size_t)(rowBlk + ra0) * DIM + k + cc);
        const int ra1 = ra0 + 64;                 // A rows 64..127
        async_copy_b128(&As[buf][ra1][cc],
                        xb + (size_t)(rowBlk + ra1) * DIM + k + cc);
    };
#if HAVE_TDM
    auto issueB = [&](int buf, int k) {
        if (wave == 0) {
            tdm_load_2d(lds_off(&Gs[buf][0][0]),
                        wgt + (size_t)colBlk * DIM + k, 64, 32, DIM);
            tdm_load_2d(lds_off(&Us[buf][0][0]),
                        wut + (size_t)colBlk * DIM + k, 64, 32, DIM);
        }
    };
#else
    auto issueB = [&](int buf, int k) {
        const int rb = wave * 8 + cr;             // B rows 0..63
        async_copy_b128(&Gs[buf][rb][cc],
                        wgt + (size_t)(colBlk + rb) * DIM + k + cc);
        async_copy_b128(&Us[buf][rb][cc],
                        wut + (size_t)(colBlk + rb) * DIM + k + cc);
    };
#endif

    v8f zero = {};
    v8f accG[2][2], accU[2][2];
#pragma unroll
    for (int m = 0; m < 2; ++m)
#pragma unroll
        for (int n = 0; n < 2; ++n) { accG[m][n] = zero; accU[m][n] = zero; }

    constexpr int NK = DIM / 32;
    issueA(0, 0);
    issueB(0, 0);
    for (int i = 0; i < NK; ++i) {
        const int buf = i & 1;
        if (i + 1 < NK) {
            issueA(buf ^ 1, (i + 1) * 32);
            issueB(buf ^ 1, (i + 1) * 32);
#if HAVE_TDM
            wait_asynccnt<2>();
            if (wave == 0) wait_tensorcnt<2>();
#else
            wait_asynccnt<4>();
#endif
        } else {
            wait_asynccnt<0>();
#if HAVE_TDM
            if (wave == 0) wait_tensorcnt<0>();
#endif
        }
        __syncthreads();

        FragAB a[2], bg[2], bu[2];
#pragma unroll
        for (int m = 0; m < 2; ++m) {
            const bf16_t* ap = &As[buf][waveM * 32 + m * 16 + llo][lhi * 16];
            a[m].q[0] = *(const v4u*)(ap);
            a[m].q[1] = *(const v4u*)(ap + 8);
        }
#pragma unroll
        for (int n = 0; n < 2; ++n) {
            const bf16_t* gp = &Gs[buf][waveN * 32 + n * 16 + llo][lhi * 16];
            bg[n].q[0] = *(const v4u*)(gp);
            bg[n].q[1] = *(const v4u*)(gp + 8);
            const bf16_t* up = &Us[buf][waveN * 32 + n * 16 + llo][lhi * 16];
            bu[n].q[0] = *(const v4u*)(up);
            bu[n].q[1] = *(const v4u*)(up + 8);
        }

#pragma unroll
        for (int m = 0; m < 2; ++m)
#pragma unroll
            for (int n = 0; n < 2; ++n) {
                accG[m][n] = __builtin_amdgcn_wmma_f32_16x16x32_bf16(
                    false, a[m].v, false, bg[n].v, (short)0, accG[m][n], false, false);
                accU[m][n] = __builtin_amdgcn_wmma_f32_16x16x32_bf16(
                    false, a[m].v, false, bu[n].v, (short)0, accU[m][n], false, false);
            }
        __syncthreads();
    }

    // Epilogue: h = silu(g)*u -> bf16. C layout: VGPR r -> row m*16 + lhi*8 + r.
#pragma unroll
    for (int m = 0; m < 2; ++m) {
        const int rowm = rowBlk + waveM * 32 + m * 16 + lhi * 8;
#pragma unroll
        for (int n = 0; n < 2; ++n) {
            const int col = colBlk + waveN * 32 + n * 16 + llo;
            v8f g = accG[m][n], u = accU[m][n];
#pragma unroll
            for (int r = 0; r < 8; ++r) {
                const float gv = g[r];
                const float hv = (gv / (1.0f + __expf(-gv))) * u[r];
                hc[(size_t)(rowm + r) * DFFN + col] = (bf16_t)hv;
            }
        }
    }
}

// ---------------------------------------------------------------------------
// Down-projection + weighted combine. A via async-to-LDS, B via TDM.
//   A = hc [T, DFF] bf16 ; B = wdt [D, DFF] bf16 ; K = DFF
//   out[t, d] (+)= w_e(t) * (A@B)[t, d],  w_0 = 1, w_e = cw[t][e-1]
// ---------------------------------------------------------------------------
__global__ __launch_bounds__(256)
void down_kernel(const bf16_t* __restrict__ hc,
                 const bf16_t* __restrict__ wdt,
                 const float* __restrict__ cw,
                 float* __restrict__ out,
                 int expert)
{
    __shared__ bf16_t As[2][128][32];
    __shared__ bf16_t Bs[2][64][32];

    const int lane  = threadIdx.x & 31;
    const int wave  = threadIdx.x >> 5;
    const int waveM = wave & 3;
    const int waveN = wave >> 2;
    const int llo   = lane & 15;
    const int lhi   = lane >> 4;

    const int rowBlk = blockIdx.y * 128;
    const int colBlk = blockIdx.x * 64;

    const int cr = lane >> 2;
    const int cc = (lane & 3) * 8;

    auto issueA = [&](int buf, int k) {
        const int ra0 = wave * 8 + cr;
        async_copy_b128(&As[buf][ra0][cc],
                        hc + (size_t)(rowBlk + ra0) * DFFN + k + cc);
        const int ra1 = ra0 + 64;
        async_copy_b128(&As[buf][ra1][cc],
                        hc + (size_t)(rowBlk + ra1) * DFFN + k + cc);
    };
#if HAVE_TDM
    auto issueB = [&](int buf, int k) {
        if (wave == 0)
            tdm_load_2d(lds_off(&Bs[buf][0][0]),
                        wdt + (size_t)colBlk * DFFN + k, 64, 32, DFFN);
    };
#else
    auto issueB = [&](int buf, int k) {
        const int rb = wave * 8 + cr;
        async_copy_b128(&Bs[buf][rb][cc],
                        wdt + (size_t)(colBlk + rb) * DFFN + k + cc);
    };
#endif

    v8f zero = {};
    v8f acc[2][2];
#pragma unroll
    for (int m = 0; m < 2; ++m)
#pragma unroll
        for (int n = 0; n < 2; ++n) acc[m][n] = zero;

    constexpr int NK = DFFN / 32;
    issueA(0, 0);
    issueB(0, 0);
    for (int i = 0; i < NK; ++i) {
        const int buf = i & 1;
        if (i + 1 < NK) {
            issueA(buf ^ 1, (i + 1) * 32);
            issueB(buf ^ 1, (i + 1) * 32);
#if HAVE_TDM
            wait_asynccnt<2>();
            if (wave == 0) wait_tensorcnt<1>();
#else
            wait_asynccnt<3>();
#endif
        } else {
            wait_asynccnt<0>();
#if HAVE_TDM
            if (wave == 0) wait_tensorcnt<0>();
#endif
        }
        __syncthreads();

        FragAB a[2], b[2];
#pragma unroll
        for (int m = 0; m < 2; ++m) {
            const bf16_t* ap = &As[buf][waveM * 32 + m * 16 + llo][lhi * 16];
            a[m].q[0] = *(const v4u*)(ap);
            a[m].q[1] = *(const v4u*)(ap + 8);
        }
#pragma unroll
        for (int n = 0; n < 2; ++n) {
            const bf16_t* bp = &Bs[buf][waveN * 32 + n * 16 + llo][lhi * 16];
            b[n].q[0] = *(const v4u*)(bp);
            b[n].q[1] = *(const v4u*)(bp + 8);
        }

#pragma unroll
        for (int m = 0; m < 2; ++m)
#pragma unroll
            for (int n = 0; n < 2; ++n)
                acc[m][n] = __builtin_amdgcn_wmma_f32_16x16x32_bf16(
                    false, a[m].v, false, b[n].v, (short)0, acc[m][n], false, false);
        __syncthreads();
    }

    // Epilogue: per-row combine weight; expert 0 overwrites, others accumulate.
#pragma unroll
    for (int m = 0; m < 2; ++m) {
        const int rowm = rowBlk + waveM * 32 + m * 16 + lhi * 8;
#pragma unroll
        for (int r = 0; r < 8; ++r) {
            const int rr = rowm + r;
            const float w = (expert == 0) ? 1.0f : cw[(size_t)rr * 7 + (expert - 1)];
#pragma unroll
            for (int n = 0; n < 2; ++n) {
                const int ccn = colBlk + waveN * 32 + n * 16 + llo;
                const size_t o = (size_t)rr * DIM + ccn;
                const float v = w * acc[m][n][r];
                out[o] = (expert == 0) ? v : (out[o] + v);
            }
        }
    }
}

// ---------------------------------------------------------------------------
extern "C" void kernel_launch(void* const* d_in, const int* in_sizes, int n_in,
                              void* d_out, int out_size, void* d_ws, size_t ws_size,
                              hipStream_t stream)
{
    (void)in_sizes; (void)n_in; (void)out_size; (void)ws_size;

    const float* x  = (const float*)d_in[0];   // [T, D]
    const float* rw = (const float*)d_in[1];   // [D, 7]
    const float* rb = (const float*)d_in[2];   // [7]
    const float* wg = (const float*)d_in[3];   // [E, D, DFF]
    const float* wu = (const float*)d_in[4];   // [E, D, DFF]
    const float* wd = (const float*)d_in[5];   // [E, DFF, D]
    float* out = (float*)d_out;                // [T, D]

    char* ws = (char*)d_ws;
    bf16_t* xb  = (bf16_t*)ws;  ws += (size_t)T_TOK * DIM  * 2;    //   8 MB
    bf16_t* wgt = (bf16_t*)ws;  ws += (size_t)NE * DFFN * DIM * 2; //  32 MB
    bf16_t* wut = (bf16_t*)ws;  ws += (size_t)NE * DFFN * DIM * 2; //  32 MB
    bf16_t* wdt = (bf16_t*)ws;  ws += (size_t)NE * DIM * DFFN * 2; //  32 MB
    bf16_t* hc  = (bf16_t*)ws;  ws += (size_t)T_TOK * DFFN * 2;    //  16 MB
    float*  cw  = (float*)ws;                                      // 112 KB

    router_kernel<<<T_TOK / 8, 256, 0, stream>>>(x, rw, rb, cw);
    convert_x_kernel<<<(T_TOK * DIM) / 256, 256, 0, stream>>>(x, xb, T_TOK * DIM);
    // wg, wu: [D, DFF] -> [DFF, D] per expert
    transpose_convert_kernel<<<dim3(DFFN / 32, DIM / 32, NE), dim3(32, 8), 0, stream>>>(wg, wgt, DIM, DFFN);
    transpose_convert_kernel<<<dim3(DFFN / 32, DIM / 32, NE), dim3(32, 8), 0, stream>>>(wu, wut, DIM, DFFN);
    // wd: [DFF, D] -> [D, DFF] per expert
    transpose_convert_kernel<<<dim3(DIM / 32, DFFN / 32, NE), dim3(32, 8), 0, stream>>>(wd, wdt, DFFN, DIM);

    for (int e = 0; e < NE; ++e) {
        gateup_kernel<<<dim3(DFFN / 64, T_TOK / 128), 256, 0, stream>>>(
            xb, wgt + (size_t)e * DFFN * DIM, wut + (size_t)e * DFFN * DIM, hc);
        down_kernel<<<dim3(DIM / 64, T_TOK / 128), 256, 0, stream>>>(
            hc, wdt + (size_t)e * DIM * DFFN, cw, out, e);
    }
}
